// SofaNet_84945863180832
// MI455X (gfx1250) — matrix operand order repaired and checked by reading, not compiled, not run on hardware
//
#include <hip/hip_runtime.h>
#include <hip/hip_bf16.h>

#define NTOT 262144
#define HDIM 128

typedef _Float16 v4h  __attribute__((ext_vector_type(4)));
typedef _Float16 v8h  __attribute__((ext_vector_type(8)));
typedef _Float16 v16h __attribute__((ext_vector_type(16)));
typedef float    v8f  __attribute__((ext_vector_type(8)));

#if defined(__gfx1250__) && __has_builtin(__builtin_amdgcn_global_load_async_to_lds_b128)
#define HAVE_ASYNC_LDS 1
#endif

#ifdef HAVE_ASYNC_LDS
typedef int v4i_gcc __attribute__((vector_size(4 * sizeof(int))));
typedef __attribute__((address_space(1))) v4i_gcc gv4i;   // global int4*
typedef __attribute__((address_space(3))) v4i_gcc lv4i;   // LDS int4*
#endif

// per-branch scalars for the _post step: {v0, dsign1} x 3 branches
__device__ float g_scal[6];

__device__ __forceinline__ v16h cat8(v8h lo, v8h hi) {
    return __builtin_shufflevector(lo, hi, 0,1,2,3,4,5,6,7,8,9,10,11,12,13,14,15);
}

__device__ __forceinline__ void async_cp_16B(const float* gsrc, float* ldst) {
#ifdef HAVE_ASYNC_LDS
    __builtin_amdgcn_global_load_async_to_lds_b128(
        (gv4i*)(__attribute__((address_space(1))) void*)(void*)const_cast<float*>(gsrc),
        (lv4i*)(__attribute__((address_space(3))) void*)(void*)ldst,
        0, 0);
#else
    (void)gsrc; (void)ldst;
#endif
}

__device__ __forceinline__ void wait_async() {
#if defined(__gfx1250__) && __has_builtin(__builtin_amdgcn_s_wait_asynccnt)
    __builtin_amdgcn_s_wait_asynccnt(0);
#elif defined(__HIP_DEVICE_COMPILE__)
    asm volatile("s_wait_asynccnt 0" ::: "memory");
#endif
}

__global__ __launch_bounds__(256) void sofa_mlp_kernel(
    const float* __restrict__ t,
    const float* __restrict__ W1, const float* __restrict__ b1,
    const float* __restrict__ W2, const float* __restrict__ b2,
    const float* __restrict__ W3, const float* __restrict__ b3,
    const float* __restrict__ W4, const float* __restrict__ b4,
    float* __restrict__ out)
{
    // f16 weights, row-major [neuron][k] == WMMA B-operand layout
    __shared__ __align__(16) _Float16 Wl[2][HDIM][HDIM];     // 64 KB
    // per-wave activation tiles: [wave][val/tan][row][k]; doubles as a 64 KB
    // f32 staging buffer for the async weight copies before layer 1 runs.
    __shared__ __align__(16) _Float16 act[8][2][16][HDIM];   // 64 KB
    __shared__ __align__(16) float bl[2][HDIM];
    __shared__ __align__(16) float w1s[HDIM], b1s[HDIM];
    __shared__ __align__(16) _Float16 w4h[HDIM];

    const int tid = threadIdx.x;
    const int br  = blockIdx.y;
    const size_t wb = (size_t)br * HDIM * HDIM;

    const int wave  = tid >> 5;
    const int lane  = tid & 31;
    const int row16 = lane & 15;
    const int hk    = lane >> 4;            // 0: lanes 0-15, 1: lanes 16-31
    const int R     = (blockIdx.x * 8 + wave) * 16;

    __builtin_prefetch(t + R, 0, 0);        // global_prefetch_b8 for the t tile

    // ---- small parameter staging ----
    if (tid < HDIM) {
        bl[0][tid] = b2[br * HDIM + tid];
        bl[1][tid] = b3[br * HDIM + tid];
        w1s[tid]   = W1[br * HDIM + tid];   // W1 is [NB][H][1]
        b1s[tid]   = b1[br * HDIM + tid];
        w4h[tid]   = (_Float16)W4[br * HDIM + tid];   // W4 is [NB][1][H]
    }

    // ---- weight staging: f32 global -> LDS (async) -> f16 Wl ----
#ifdef HAVE_ASYNC_LDS
    {
        float* stage = reinterpret_cast<float*>(&act[0][0][0][0]);  // 64 KB
        for (int pass = 0; pass < 2; ++pass) {
            const float* src = (pass == 0 ? W2 : W3) + wb;
#pragma unroll
            for (int i = 0; i < 16; ++i) {
                const int idx4 = tid + i * 256;               // float4 index
                async_cp_16B(src + idx4 * 4, stage + idx4 * 4);
            }
            wait_async();
            __syncthreads();
            const float4* s4 = reinterpret_cast<const float4*>(stage);
            v4h* d4 = reinterpret_cast<v4h*>(&Wl[pass][0][0]);
            for (int i = tid; i < HDIM * HDIM / 4; i += 256) {
                float4 a = s4[i];
                v4h h = { (_Float16)a.x, (_Float16)a.y,
                          (_Float16)a.z, (_Float16)a.w };
                d4[i] = h;
            }
            __syncthreads();
        }
    }
#else
    for (int i = tid; i < HDIM * HDIM / 4; i += 256) {
        float4 a = reinterpret_cast<const float4*>(W2 + wb)[i];
        float4 c = reinterpret_cast<const float4*>(W3 + wb)[i];
        v4h h2 = { (_Float16)a.x, (_Float16)a.y, (_Float16)a.z, (_Float16)a.w };
        v4h h3 = { (_Float16)c.x, (_Float16)c.y, (_Float16)c.z, (_Float16)c.w };
        reinterpret_cast<v4h*>(&Wl[0][0][0])[i] = h2;
        reinterpret_cast<v4h*>(&Wl[1][0][0])[i] = h3;
    }
    __syncthreads();
#endif

    // ================= layer 1: elementwise 1 -> 128 (+ tangent) ===========
    {
        float  tv = t[R + row16];           // lane L holds t[R + (L&15)]
        float4 wv = reinterpret_cast<const float4*>(w1s)[lane];
        float4 bv = reinterpret_cast<const float4*>(b1s)[lane];
#pragma unroll
        for (int r = 0; r < 16; ++r) {
            float tr = __shfl(tv, r, 32);
            float z0 = tr * wv.x + bv.x;
            float z1 = tr * wv.y + bv.y;
            float z2 = tr * wv.z + bv.z;
            float z3 = tr * wv.w + bv.w;
            v4h hv = { (_Float16)fmaxf(z0, 0.f), (_Float16)fmaxf(z1, 0.f),
                       (_Float16)fmaxf(z2, 0.f), (_Float16)fmaxf(z3, 0.f) };
            v4h gv = { (_Float16)(z0 > 0.f ? wv.x : 0.f),
                       (_Float16)(z1 > 0.f ? wv.y : 0.f),
                       (_Float16)(z2 > 0.f ? wv.z : 0.f),
                       (_Float16)(z3 > 0.f ? wv.w : 0.f) };
            reinterpret_cast<v4h*>(&act[wave][0][r][0])[lane] = hv;
            reinterpret_cast<v4h*>(&act[wave][1][r][0])[lane] = gv;
        }
    }

    // ================= layers 2 & 3: 128x128 via WMMA (value + tangent) ====
    for (int L = 0; L < 2; ++L) {
        // Hoist all A-operands (value & tangent, full K) into registers so we
        // can update the activation buffer in place.
        v16h Av[4], At[4];
        {
            const v8h* pv = reinterpret_cast<const v8h*>(&act[wave][0][row16][0]);
            const v8h* pt = reinterpret_cast<const v8h*>(&act[wave][1][row16][0]);
#pragma unroll
            for (int kk = 0; kk < 4; ++kk) {
                Av[kk] = cat8(pv[kk * 4 + hk], pv[kk * 4 + 2 + hk]);
                At[kk] = cat8(pt[kk * 4 + hk], pt[kk * 4 + 2 + hk]);
            }
        }
#pragma unroll
        for (int nn = 0; nn < 8; ++nn) {
            const int col = nn * 16 + row16;          // output neuron per lane
            v8f cv = {};
            v8f ct = {};
            const v8h* pb = reinterpret_cast<const v8h*>(&Wl[L][col][0]);
#pragma unroll
            for (int kk = 0; kk < 4; ++kk) {
                v16h B = cat8(pb[kk * 4 + hk], pb[kk * 4 + 2 + hk]);
                cv = __builtin_amdgcn_wmma_f32_16x16x32_f16(
                        false, Av[kk], false, B, (short)0, cv, false, false);
                ct = __builtin_amdgcn_wmma_f32_16x16x32_f16(
                        false, At[kk], false, B, (short)0, ct, false, false);
            }
            const float bc = bl[L][col];
#pragma unroll
            for (int i = 0; i < 8; ++i) {
                float z  = cv[i] + bc;
                int   rr = i + hk * 8;                // C/D VGPR layout row
                act[wave][0][rr][col] = (_Float16)(z > 0.f ? z     : 0.f);
                act[wave][1][rr][col] = (_Float16)(z > 0.f ? ct[i] : 0.f);
            }
        }
    }

    // ======== layer 4 as WMMA: B has w4 in column 0, zeros elsewhere =======
    {
        v16h Av[4], At[4];
        {
            const v8h* pv = reinterpret_cast<const v8h*>(&act[wave][0][row16][0]);
            const v8h* pt = reinterpret_cast<const v8h*>(&act[wave][1][row16][0]);
#pragma unroll
            for (int kk = 0; kk < 4; ++kk) {
                Av[kk] = cat8(pv[kk * 4 + hk], pv[kk * 4 + 2 + hk]);
                At[kk] = cat8(pt[kk * 4 + hk], pt[kk * 4 + 2 + hk]);
            }
        }
        v8f c4v = {};
        v8f c4t = {};
        const v8h* pw = reinterpret_cast<const v8h*>(w4h);
#pragma unroll
        for (int kk = 0; kk < 4; ++kk) {
            v8h lo = {};
            v8h hi = {};
            if (row16 == 0) {                     // only column 0 is live
                lo = pw[kk * 4 + hk];
                hi = pw[kk * 4 + 2 + hk];
            }
            v16h B = cat8(lo, hi);
            c4v = __builtin_amdgcn_wmma_f32_16x16x32_f16(
                    false, Av[kk], false, B, (short)0, c4v, false, false);
            c4t = __builtin_amdgcn_wmma_f32_16x16x32_f16(
                    false, At[kk], false, B, (short)0, c4t, false, false);
        }
        if (row16 == 0) {                         // lanes 0 (rows 0-7) and 16 (rows 8-15)
            const float bias4 = b4[br];
            const int rbase = R + hk * 8;
#pragma unroll
            for (int i = 0; i < 8; ++i) {
                out[(size_t)br * NTOT + rbase + i]       = c4v[i] + bias4;
                out[(size_t)(3 + br) * NTOT + rbase + i] = c4t[i];
            }
        }
    }
}

// Extract the cross-element scalars needed by _post (v[0], sign at index 1).
__global__ void sofa_scal_kernel(const float* __restrict__ out)
{
    int b = threadIdx.x;
    if (b < 3) {
        float v0 = out[(size_t)b * NTOT];
        float v1 = out[(size_t)b * NTOT + 1];
        float d1 = out[(size_t)(3 + b) * NTOT + 1];
        float s  = (b == 1) ? -1.f : 1.f;
        float df = v1 - v0;
        float sg = (df > 0.f) ? 1.f : ((df < 0.f) ? -1.f : 0.f);
        g_scal[b * 2]     = v0;
        g_scal[b * 2 + 1] = s * sg * d1;
    }
}

// Elementwise _post, in place over d_out (each thread touches only its own 2).
__global__ __launch_bounds__(256) void sofa_post_kernel(float* __restrict__ out)
{
    int i = blockIdx.x * 256 + threadIdx.x;
    int b = blockIdx.y;
    float v  = out[(size_t)b * NTOT + i];
    float dt = out[(size_t)(3 + b) * NTOT + i];
    float s  = (b == 1) ? -1.f : 1.f;
    float v0 = g_scal[b * 2];
    float d1 = g_scal[b * 2 + 1];
    float df = v - v0;
    float sg = (df > 0.f) ? 1.f : ((df < 0.f) ? -1.f : 0.f);
    float vv = s * fabsf(df);
    float ds = s * sg * dt;
    if (i == 0) ds = (d1 >= 0.f) ? fabsf(dt) : -fabsf(dt);
    out[(size_t)b * NTOT + i]       = vv;
    out[(size_t)(3 + b) * NTOT + i] = ds;
}

extern "C" void kernel_launch(void* const* d_in, const int* in_sizes, int n_in,
                              void* d_out, int out_size, void* d_ws, size_t ws_size,
                              hipStream_t stream)
{
    (void)in_sizes; (void)n_in; (void)out_size; (void)d_ws; (void)ws_size;
    const float* t  = (const float*)d_in[0];
    const float* W1 = (const float*)d_in[1];
    const float* b1 = (const float*)d_in[2];
    const float* W2 = (const float*)d_in[3];
    const float* b2 = (const float*)d_in[4];
    const float* W3 = (const float*)d_in[5];
    const float* b3 = (const float*)d_in[6];
    const float* W4 = (const float*)d_in[7];
    const float* b4 = (const float*)d_in[8];
    float* out = (float*)d_out;

    sofa_mlp_kernel<<<dim3(NTOT / 128, 3), 256, 0, stream>>>(
        t, W1, b1, W2, b2, W3, b3, W4, b4, out);
    sofa_scal_kernel<<<dim3(1), dim3(4), 0, stream>>>(out);
    sofa_post_kernel<<<dim3(NTOT / 256, 3), 256, 0, stream>>>(out);
}
